// AttentionConv_34454227648934
// MI455X (gfx1250) — compile-verified, matrix operand order checked by reading, not run on hardware
//
#include <hip/hip_runtime.h>

#define DIM 384
#define HQ 3136          // q tokens per batch (56*56)
#define HK 784           // k/v tokens per batch (28*28)
#define TPAD 800         // 784 padded to multiple of 32
#define BATCH 8
#define NHEAD 6
#define HD 64
#define SCALE 0.05103103630798288f   // 384^-0.5

typedef __attribute__((ext_vector_type(16))) __bf16 v16bf;
typedef __attribute__((ext_vector_type(8)))  __bf16 v8bf;
typedef __attribute__((ext_vector_type(8)))  float  v8f;

__device__ __forceinline__ unsigned short f2bs(float f) {
    unsigned int u = __float_as_uint(f);
    u += 0x7fffu + ((u >> 16) & 1u);   // round to nearest even
    return (unsigned short)(u >> 16);
}
__device__ __forceinline__ __bf16 f2b(float f) {
    return __builtin_bit_cast(__bf16, f2bs(f));
}

// Load a 16-element bf16 fragment: two contiguous 8-elem K-chunks at +0 and +16
__device__ __forceinline__ v16bf load_frag(const unsigned short* p) {
    v8bf lo = *(const v8bf*)p;
    v8bf hi = *(const v8bf*)(p + 16);
    return __builtin_shufflevector(lo, hi, 0,1,2,3,4,5,6,7,8,9,10,11,12,13,14,15);
}

__device__ __forceinline__ v8f wmma_bf(v16bf a, v16bf b, v8f c) {
    return __builtin_amdgcn_wmma_f32_16x16x32_bf16(false, a, false, b, (short)0, c, false, false);
}

// ---------------- weight / misc conversion kernels ----------------
__global__ void cvt_bf16_kernel(const float* __restrict__ in,
                                unsigned short* __restrict__ out, int n) {
    int i = blockIdx.x * 256 + threadIdx.x;
    if (i < n) out[i] = f2bs(in[i]);
}

__global__ void zero_vt_pad_kernel(unsigned short* __restrict__ vt) {
    // zero t in [784,800) for every (b,h,d) row so padded K-steps contribute 0
    int i = blockIdx.x * 256 + threadIdx.x;
    int nrows = BATCH * NHEAD * HD;
    if (i < nrows * 16) {
        int row = i >> 4, t = HK + (i & 15);
        vt[(size_t)row * TPAD + t] = 0;  // +0.0 bf16
    }
}

// ---------------- depthwise 3x3 conv + BN -> bf16, [B, Ho*Wo, C] ----------------
// outScale folds the attention softmax scale into the q branch (linear chain).
__global__ void __launch_bounds__(DIM)
dwconv_bn_kernel(const float* __restrict__ x, const float* __restrict__ w,
                 const float* __restrict__ g, const float* __restrict__ bb,
                 const float* __restrict__ mm, const float* __restrict__ vv,
                 unsigned short* __restrict__ out, int stride, int Ho, int Wo,
                 float outScale) {
    const int c = threadIdx.x;
    const int pix = blockIdx.x;
    const int b  = pix / (Ho * Wo);
    const int rm = pix % (Ho * Wo);
    const int oy = rm / Wo, ox = rm % Wo;
    const int iy0 = oy * stride - 1, ix0 = ox * stride - 1;
    float acc = 0.f;
#pragma unroll
    for (int ky = 0; ky < 3; ++ky) {
        int iy = iy0 + ky;
        if (iy < 0 || iy >= 56) continue;
#pragma unroll
        for (int kx = 0; kx < 3; ++kx) {
            int ix = ix0 + kx;
            if (ix < 0 || ix >= 56) continue;
            acc += x[((size_t)(b * 56 + iy) * 56 + ix) * DIM + c] * w[c * 9 + ky * 3 + kx];
        }
    }
    float inv = rsqrtf(vv[c] + 1e-5f);
    float y = (acc - mm[c]) * (inv * g[c]) + bb[c];
    out[(size_t)pix * DIM + c] = f2bs(y * outScale);
}

// ---------------- generic 384-K GEMM: out[m, o] = A[m,:] . W[o,:] ----------------
// Software-pipelined: next K-step fragments are loaded before current WMMAs.
// MODE 0: bf16 store [M,384];  MODE 1: bf16 store transposed per head into vT[b][h][d][TPAD];
// MODE 2: f32 store [M,384] + bias
template <int MODE>
__global__ void __launch_bounds__(128)
gemm384_kernel(const unsigned short* __restrict__ A, const unsigned short* __restrict__ W,
               unsigned short* __restrict__ outb, float* __restrict__ outf,
               const float* __restrict__ bias) {
    const int lane = threadIdx.x & 31;
    const int wg = blockIdx.x * 4 + (threadIdx.x >> 5);
    const int mt = wg / 6, nq = wg % 6;
    const int row0 = mt * 16, col0 = nq * 64;
    const int r = lane & 15, ko = (lane >> 4) * 8;

    v8f acc0 = {}, acc1 = {}, acc2 = {}, acc3 = {};
    const unsigned short* ap = A + (size_t)(row0 + r) * DIM + ko;
    const unsigned short* wp = W + (size_t)(col0 + r) * DIM + ko;

    v16bf af  = load_frag(ap);
    v16bf bf0 = load_frag(wp);
    v16bf bf1 = load_frag(wp + 16 * DIM);
    v16bf bf2 = load_frag(wp + 32 * DIM);
    v16bf bf3 = load_frag(wp + 48 * DIM);
#pragma unroll
    for (int k0 = 0; k0 < DIM; k0 += 32) {
        v16bf naf = af, nb0 = bf0, nb1 = bf1, nb2 = bf2, nb3 = bf3;
        if (k0 + 32 < DIM) {   // static under full unroll
            naf = load_frag(ap + k0 + 32);
            nb0 = load_frag(wp + k0 + 32);
            nb1 = load_frag(wp + 16 * DIM + k0 + 32);
            nb2 = load_frag(wp + 32 * DIM + k0 + 32);
            nb3 = load_frag(wp + 48 * DIM + k0 + 32);
        }
        acc0 = wmma_bf(af, bf0, acc0);
        acc1 = wmma_bf(af, bf1, acc1);
        acc2 = wmma_bf(af, bf2, acc2);
        acc3 = wmma_bf(af, bf3, acc3);
        af = naf; bf0 = nb0; bf1 = nb1; bf2 = nb2; bf3 = nb3;
    }
    v8f acc[4] = {acc0, acc1, acc2, acc3};

    const int rowb = row0 + (lane >> 4) * 8;
    const int colb = lane & 15;
#pragma unroll
    for (int n = 0; n < 4; ++n) {
#pragma unroll
        for (int j = 0; j < 8; ++j) {
            int row = rowb + j;
            int col = col0 + n * 16 + colb;
            float v = acc[n][j];
            if (MODE == 0) {
                outb[(size_t)row * DIM + col] = f2bs(v);
            } else if (MODE == 1) {
                int b = row / HK, t = row % HK;
                int h = col >> 6, d = col & 63;
                outb[(((size_t)b * NHEAD + h) * HD + d) * TPAD + t] = f2bs(v);
            } else {
                outf[(size_t)row * DIM + col] = v + bias[col];
            }
        }
    }
}

// ---------------- attention: per wave one 16-query tile of one (b,h) ----------------
__global__ void __launch_bounds__(64)
attn_kernel(const unsigned short* __restrict__ Q, const unsigned short* __restrict__ K,
            const unsigned short* __restrict__ VT, unsigned short* __restrict__ O) {
    extern __shared__ float smem[];
    const int lane = threadIdx.x & 31;
    const int wv = threadIdx.x >> 5;
    float* P = smem + (size_t)wv * 16 * TPAD;

    const int g = blockIdx.x * 2 + wv;
    const int qt = g % 196;
    const int bh = g / 196;
    const int b = bh / NHEAD, h = bh % NHEAD;

    const int r = lane & 15, ko = (lane >> 4) * 8;
    const size_t qrow0 = (size_t)b * HQ + qt * 16;
    const size_t krow0 = (size_t)b * HK;
    const int hc = h * HD;

    // ---- phase 1: S = (q*scale) Kᵀ -> LDS (scale pre-folded into qc) ----
    const unsigned short* qp = Q + (qrow0 + r) * DIM + hc + ko;
    v16bf aq0 = load_frag(qp);
    v16bf aq1 = load_frag(qp + 32);
    const unsigned short* kp = K + (krow0 + r) * DIM + hc + ko;
    v16bf kf0 = load_frag(kp);
    v16bf kf1 = load_frag(kp + 32);
    const int prow = (lane >> 4) * 8;
    for (int tt = 0; tt < 49; ++tt) {
        v16bf n0 = kf0, n1 = kf1;
        if (tt < 48) {
            const unsigned short* np = kp + (size_t)(tt + 1) * 16 * DIM;
            n0 = load_frag(np);
            n1 = load_frag(np + 32);
            if (tt < 47)
                __builtin_prefetch(kp + (size_t)(tt + 2) * 16 * DIM, 0, 1);
        }
        v8f acc = {};
        acc = wmma_bf(aq0, kf0, acc);
        acc = wmma_bf(aq1, kf1, acc);
        int t0 = tt * 16;
#pragma unroll
        for (int j = 0; j < 8; ++j)
            P[(prow + j) * TPAD + t0 + (lane & 15)] = acc[j];
        kf0 = n0; kf1 = n1;
    }
    {   // zero pad columns t in [784,800)
        int row = lane & 15, st = HK + (lane >> 4) * 8;
#pragma unroll
        for (int i = 0; i < 8; ++i) P[row * TPAD + st + i] = 0.f;
    }
    __syncthreads();

    // ---- phase 2: softmax (each row handled by a lane pair via shfl_xor 16) ----
    float* Pr = P + (lane & 15) * TPAD + (lane >> 4) * 392;
    float mx = -3.0e38f;
    for (int i = 0; i < 392; ++i) mx = fmaxf(mx, Pr[i]);
    mx = fmaxf(mx, __shfl_xor(mx, 16, 32));
    float sum = 0.f;
    for (int i = 0; i < 392; ++i) { float e = __expf(Pr[i] - mx); Pr[i] = e; sum += e; }
    sum += __shfl_xor(sum, 16, 32);
    float rinv = 1.0f / sum;
    __syncthreads();

    // ---- phase 3: O = P V (K axis = t, vT gives K-contiguous B frags) ----
    v8f acc[4] = {{}, {}, {}, {}};
    const unsigned short* vbase = VT + (((size_t)b * NHEAD + h) * HD) * TPAD
                                + (size_t)(lane & 15) * TPAD + ko;
    v16bf vf0 = load_frag(vbase);
    v16bf vf1 = load_frag(vbase + 16 * TPAD);
    v16bf vf2 = load_frag(vbase + 32 * TPAD);
    v16bf vf3 = load_frag(vbase + 48 * TPAD);
    for (int ts = 0; ts < 25; ++ts) {
        int t0 = ts * 32;
        v16bf n0 = vf0, n1 = vf1, n2 = vf2, n3 = vf3;
        if (ts < 24) {
            n0 = load_frag(vbase + t0 + 32);
            n1 = load_frag(vbase + 16 * TPAD + t0 + 32);
            n2 = load_frag(vbase + 32 * TPAD + t0 + 32);
            n3 = load_frag(vbase + 48 * TPAD + t0 + 32);
        }
        const float* pa = P + (lane & 15) * TPAD + t0 + ko;
        v8f lof = *(const v8f*)pa;
        v8f hif = *(const v8f*)(pa + 16);
        v16bf af;
#pragma unroll
        for (int i = 0; i < 8; ++i) { af[i] = f2b(lof[i]); af[i + 8] = f2b(hif[i]); }
        acc[0] = wmma_bf(af, vf0, acc[0]);
        acc[1] = wmma_bf(af, vf1, acc[1]);
        acc[2] = wmma_bf(af, vf2, acc[2]);
        acc[3] = wmma_bf(af, vf3, acc[3]);
        vf0 = n0; vf1 = n1; vf2 = n2; vf3 = n3;
    }
    // store, folding 1/sum (row stat lives in lane == row, 0..15)
#pragma unroll
    for (int n = 0; n < 4; ++n) {
#pragma unroll
        for (int j = 0; j < 8; ++j) {
            int rrow = prow + j;
            float rv = __shfl(rinv, rrow, 32);
            O[(qrow0 + rrow) * DIM + hc + n * 16 + (lane & 15)] = f2bs(acc[n][j] * rv);
        }
    }
}

// ---------------- host launcher ----------------
extern "C" void kernel_launch(void* const* d_in, const int* in_sizes, int n_in,
                              void* d_out, int out_size, void* d_ws, size_t ws_size,
                              hipStream_t stream) {
    (void)in_sizes; (void)n_in; (void)out_size; (void)ws_size;
    const float* x      = (const float*)d_in[0];
    const float* conv_q = (const float*)d_in[3];
    const float* bnq_s  = (const float*)d_in[4];
    const float* bnq_b  = (const float*)d_in[5];
    const float* bnq_m  = (const float*)d_in[6];
    const float* bnq_v  = (const float*)d_in[7];
    const float* conv_k = (const float*)d_in[8];
    const float* bnk_s  = (const float*)d_in[9];
    const float* bnk_b  = (const float*)d_in[10];
    const float* bnk_m  = (const float*)d_in[11];
    const float* bnk_v  = (const float*)d_in[12];
    const float* conv_v = (const float*)d_in[13];
    const float* bnv_s  = (const float*)d_in[14];
    const float* bnv_b  = (const float*)d_in[15];
    const float* bnv_m  = (const float*)d_in[16];
    const float* bnv_v  = (const float*)d_in[17];
    const float* wq     = (const float*)d_in[18];
    const float* wk     = (const float*)d_in[19];
    const float* wvv    = (const float*)d_in[20];
    const float* wl     = (const float*)d_in[21];
    const float* b_last = (const float*)d_in[22];
    float* out = (float*)d_out;

    char* ws = (char*)d_ws;
    size_t off = 0;
    auto take = [&](size_t bytes) -> char* {
        char* p = ws + off;
        off = (off + bytes + 255) & ~(size_t)255;
        return p;
    };
    const size_t MQ = (size_t)BATCH * HQ;  // 25088
    const size_t MK = (size_t)BATCH * HK;  // 6272
    unsigned short* qc  = (unsigned short*)take(MQ * DIM * 2);
    unsigned short* kc  = (unsigned short*)take(MK * DIM * 2);
    unsigned short* vc  = (unsigned short*)take(MK * DIM * 2);
    unsigned short* wqb = (unsigned short*)take((size_t)DIM * DIM * 2);
    unsigned short* wkb = (unsigned short*)take((size_t)DIM * DIM * 2);
    unsigned short* wvb = (unsigned short*)take((size_t)DIM * DIM * 2);
    unsigned short* wlb = (unsigned short*)take((size_t)DIM * DIM * 2);
    unsigned short* Qp  = (unsigned short*)take(MQ * DIM * 2);
    unsigned short* Kp  = (unsigned short*)take(MK * DIM * 2);
    unsigned short* vT  = (unsigned short*)take((size_t)BATCH * NHEAD * HD * TPAD * 2);
    unsigned short* Ob  = (unsigned short*)take(MQ * DIM * 2);

    const int NW = DIM * DIM;  // 147456
    cvt_bf16_kernel<<<dim3((NW + 255) / 256), dim3(256), 0, stream>>>(wq,  wqb, NW);
    cvt_bf16_kernel<<<dim3((NW + 255) / 256), dim3(256), 0, stream>>>(wk,  wkb, NW);
    cvt_bf16_kernel<<<dim3((NW + 255) / 256), dim3(256), 0, stream>>>(wvv, wvb, NW);
    cvt_bf16_kernel<<<dim3((NW + 255) / 256), dim3(256), 0, stream>>>(wl,  wlb, NW);
    zero_vt_pad_kernel<<<dim3((BATCH * NHEAD * HD * 16 + 255) / 256), dim3(256), 0, stream>>>(vT);

    dwconv_bn_kernel<<<dim3((unsigned)MQ), dim3(DIM), 0, stream>>>(
        x, conv_q, bnq_s, bnq_b, bnq_m, bnq_v, qc, 1, 56, 56, SCALE);
    dwconv_bn_kernel<<<dim3((unsigned)MK), dim3(DIM), 0, stream>>>(
        x, conv_k, bnk_s, bnk_b, bnk_m, bnk_v, kc, 2, 28, 28, 1.0f);
    dwconv_bn_kernel<<<dim3((unsigned)MK), dim3(DIM), 0, stream>>>(
        x, conv_v, bnv_s, bnv_b, bnv_m, bnv_v, vc, 2, 28, 28, 1.0f);

    // projections: waves = (M/16)*6, 4 waves/block
    gemm384_kernel<0><<<dim3((unsigned)(MQ / 16 * 6 / 4)), dim3(128), 0, stream>>>(
        qc, wqb, Qp, nullptr, nullptr);
    gemm384_kernel<0><<<dim3((unsigned)(MK / 16 * 6 / 4)), dim3(128), 0, stream>>>(
        kc, wkb, Kp, nullptr, nullptr);
    gemm384_kernel<1><<<dim3((unsigned)(MK / 16 * 6 / 4)), dim3(128), 0, stream>>>(
        vc, wvb, vT, nullptr, nullptr);

    // attention: 8*6*196 q-tiles, 2 waves/block, 2*16*800*4 bytes LDS
    attn_kernel<<<dim3(BATCH * NHEAD * 196 / 2), dim3(64), 2 * 16 * TPAD * 4, stream>>>(
        Qp, Kp, vT, Ob);

    // output projection + bias (f32 out)
    gemm384_kernel<2><<<dim3((unsigned)(MQ / 16 * 6 / 4)), dim3(128), 0, stream>>>(
        Ob, wlb, nullptr, out, b_last);
}
